// CausalSelfAttention_72730976190491
// MI455X (gfx1250) — matrix operand order repaired and checked by reading, not compiled, not run on hardware
//
#include <hip/hip_runtime.h>
#include <hip/hip_bf16.h>

// ---------------------------------------------------------------------------
// CausalSelfAttention for MI455X (gfx1250): bf16 WMMA pipeline
//   1) convert x / qkv_w / out_w to bf16 in workspace
//   2) QKV GEMM (WMMA bf16, fp32 acc) -> head-major bf16 [3,B,H,T,D]
//   3) RoPE applied in place to q,k
//   4) flash attention (online softmax, WMMA for QK^T and P*V,
//      K tile staged to LDS by the Tensor Data Mover)
//   5) output projection GEMM (WMMA) + bias -> fp32 d_out
// ---------------------------------------------------------------------------

#define B_   4
#define T_   2048
#define C_   1024
#define H_   16
#define D_   64
#define K3C  3072   // 3*C
#define BT   8192   // B*T

typedef __bf16 bf16_t;
typedef __attribute__((ext_vector_type(16))) __bf16        v16bf;
typedef __attribute__((ext_vector_type(8)))  float         v8f;
typedef __attribute__((ext_vector_type(4)))  unsigned int  v4u;
typedef __attribute__((ext_vector_type(4)))  unsigned int  u32x4;
typedef __attribute__((ext_vector_type(8)))  int           i32x8;
typedef __attribute__((ext_vector_type(4)))  int           i32x4;

#if defined(__has_builtin)
#  if __has_builtin(__builtin_amdgcn_tensor_load_to_lds) && \
      __has_builtin(__builtin_amdgcn_s_wait_tensorcnt)
#    define USE_TDM 1
#  endif
#endif

struct U32x8 { v4u lo, hi; };

static __device__ __forceinline__ v16bf make_frag(const unsigned short* p0,
                                                  const unsigned short* p1) {
  U32x8 u;
  u.lo = *(const v4u*)p0;
  u.hi = *(const v4u*)p1;
  return __builtin_bit_cast(v16bf, u);
}

static __device__ __forceinline__ unsigned short f2bf(float f) {
  unsigned int u = __builtin_bit_cast(unsigned int, f);
  unsigned int r = u + 0x7FFFu + ((u >> 16) & 1u);  // round-to-nearest-even
  return (unsigned short)(r >> 16);
}

static __device__ __forceinline__ float bf2f(unsigned short h) {
  unsigned int u = ((unsigned int)h) << 16;
  return __builtin_bit_cast(float, u);
}

static __device__ __forceinline__ v8f wmma_bf16(v16bf a, v16bf b, v8f c) {
  // D = A(16x32 bf16) * B(32x16 bf16) + C(16x16 f32)
  return __builtin_amdgcn_wmma_f32_16x16x32_bf16(
      false, a, false, b, (short)0, c, false, false);
}

// ---------------------------------------------------------------------------
// fp32 -> bf16 conversion
// ---------------------------------------------------------------------------
__global__ void cvt_f32_bf16(const float* __restrict__ in,
                             unsigned short* __restrict__ out, int n) {
  int i = blockIdx.x * blockDim.x + threadIdx.x;
  int stride = gridDim.x * blockDim.x;
  for (; i < n; i += stride) out[i] = f2bf(in[i]);
}

// ---------------------------------------------------------------------------
// QKV GEMM: [8192 x 1024] x [3072 x 1024]^T, bias add, scatter to
// qkvb[sel][b][h][t][d] (bf16). Block tile 128x128, 8 waves of 32x64
// (2x4 WMMA accumulators -> 8 v_wmma per K-step).
// ---------------------------------------------------------------------------
__global__ __launch_bounds__(256) void qkv_gemm(
    const unsigned short* __restrict__ xb,     // [BT][C] bf16
    const unsigned short* __restrict__ wb,     // [3C][C] bf16
    const float* __restrict__ bias,            // [3C]
    unsigned short* __restrict__ qkvb) {       // [3][B][H][T][D] bf16
  const int block_m = blockIdx.x * 128;
  const int block_n = blockIdx.y * 128;
  const int wave = threadIdx.x >> 5;
  const int lane = threadIdx.x & 31;
  const int lr = lane & 15;
  const int lh = (lane < 16) ? 0 : 1;
  const int m0 = block_m + (wave >> 1) * 32;   // 4 row-waves
  const int n0 = block_n + (wave & 1) * 64;    // 2 col-waves

  v8f acc[2][4] = {};
  for (int k0 = 0; k0 < C_; k0 += 32) {
    v16bf afr[2], bfr[4];
    for (int mi = 0; mi < 2; ++mi) {
      const unsigned short* ap = xb + (size_t)(m0 + mi * 16 + lr) * C_ + k0 + lh * 8;
      __builtin_prefetch(ap + 64, 0, 1);   // global_prefetch_b8: next K tiles
      afr[mi] = make_frag(ap, ap + 16);
    }
    for (int ni = 0; ni < 4; ++ni) {
      const unsigned short* bp = wb + (size_t)(n0 + ni * 16 + lr) * C_ + k0 + lh * 8;
      bfr[ni] = make_frag(bp, bp + 16);
    }
    for (int mi = 0; mi < 2; ++mi)
      for (int ni = 0; ni < 4; ++ni)
        acc[mi][ni] = wmma_bf16(afr[mi], bfr[ni], acc[mi][ni]);
  }
  // epilogue: bias + bf16 + scatter to [3][B][H][T][D]
  for (int mi = 0; mi < 2; ++mi) {
    for (int ni = 0; ni < 4; ++ni) {
      int gn = n0 + ni * 16 + lr;          // 0..3071
      int sel = gn >> 10;
      int h = (gn & 1023) >> 6;
      int d = gn & 63;
      float bia = bias[gn];
      for (int r = 0; r < 8; ++r) {
        int gm = m0 + mi * 16 + r + lh * 8;  // B*T row
        int b = gm >> 11;
        int t = gm & (T_ - 1);
        size_t dst = ((((size_t)sel * B_ + b) * H_ + h) * T_ + t) * D_ + d;
        qkvb[dst] = f2bf(acc[mi][ni][r] + bia);
      }
    }
  }
}

// ---------------------------------------------------------------------------
// RoPE in place on q (sel 0) and k (sel 1). One thread per (row, pair).
// ---------------------------------------------------------------------------
__global__ void rope_kernel(unsigned short* __restrict__ qkvb) {
  const int total = 2 * B_ * H_ * T_ * (D_ / 2);  // 8388608
  int idx = blockIdx.x * blockDim.x + threadIdx.x;
  if (idx >= total) return;
  int p = idx & 31;           // pair index d/2
  int rest = idx >> 5;        // (sel,b,h,t) flat row
  int t = rest & (T_ - 1);
  size_t base = (size_t)rest * D_ + p * 2;
  float theta = (float)t * __powf(10000.0f, -(float)(2 * p) / (float)D_);
  float s, c;
  __sincosf(theta, &s, &c);
  float xe = bf2f(qkvb[base]);
  float xo = bf2f(qkvb[base + 1]);
  qkvb[base]     = f2bf(xe * c - xo * s);
  qkvb[base + 1] = f2bf(xe * s + xo * c);
}

// ---------------------------------------------------------------------------
// Flash attention. grid = (T/128, B*H), block = 256 (8 waves).
// Each wave: 16 query rows, online softmax. K tile staged via TDM,
// V tile transposed into LDS manually.
// ---------------------------------------------------------------------------
__global__ __launch_bounds__(256) void flash_attn(
    const unsigned short* __restrict__ qkvb,  // [3][B][H][T][D] bf16 (rope'd)
    unsigned short* __restrict__ attnb) {     // [B][T][H*D] bf16
  __shared__ unsigned short Ks[32][64];       // [key][d]
  __shared__ unsigned short Vt[64][32];       // [d][key] (transposed)
  __shared__ unsigned short Ps[8][16][32];    // per-wave P tile [row][key]

  const int wave = threadIdx.x >> 5;
  const int lane = threadIdx.x & 31;
  const int lr = lane & 15;
  const int lh = (lane < 16) ? 0 : 1;
  const int b = blockIdx.y >> 4;
  const int h = blockIdx.y & 15;
  const int qblock = blockIdx.x * 128;
  const int q0 = qblock + wave * 16;

  const size_t head = (size_t)(b * H_ + h) * T_ * D_;
  const unsigned short* Qg = qkvb + head;
  const unsigned short* Kg = qkvb + (size_t)B_ * H_ * T_ * D_ + head;
  const unsigned short* Vg = qkvb + (size_t)2 * B_ * H_ * T_ * D_ + head;

  // Q fragments (A layout), loaded once
  const unsigned short* Qrow = Qg + (size_t)(q0 + lr) * D_;
  v16bf qf[2];
  for (int j = 0; j < 2; ++j) {
    const unsigned short* ap = Qrow + j * 32 + lh * 8;
    qf[j] = make_frag(ap, ap + 16);
  }

  v8f o[4] = {};
  float mrow[8], lrow[8];
  for (int r = 0; r < 8; ++r) { mrow[r] = -1e30f; lrow[r] = 0.0f; }

  const int nkt = blockIdx.x * 4 + 4;  // key tiles of 32 up to block's max row
  const int tid = threadIdx.x;
  const int srow = tid >> 3;           // 0..31
  const int scol = (tid & 7) * 8;      // 0..56 step 8

  for (int kt = 0; kt < nkt; ++kt) {
    // ---- stage V tile: load rows, scatter-transpose into Vt ----
    {
      union { v4u v; unsigned short s[8]; } u;
      u.v = *(const v4u*)(Vg + (size_t)(kt * 32 + srow) * D_ + scol);
      for (int j = 0; j < 8; ++j) Vt[scol + j][srow] = u.s[j];
    }
    // ---- stage K tile (32 keys x 64 d, row-major) ----
#if defined(USE_TDM)
    if (wave == 0) {
      // Tensor DMA descriptor (D#): 2D tile, 2-byte elements,
      // tile_dim0=64 (d), tile_dim1=32 (keys), stride 64 elements.
      unsigned long long ga =
          (unsigned long long)(uintptr_t)(Kg + (size_t)kt * 32 * D_);
      u32x4 g0 = {
        1u,                                        // count=1 (valid user D#)
        (unsigned)(uintptr_t)&Ks[0][0],            // lds_addr
        (unsigned)(ga & 0xFFFFFFFFu),              // global_addr[31:0]
        (unsigned)((ga >> 32) & 0x1FFFFFFu) | (2u << 30)  // addr[56:32], type=2
      };
      i32x8 g1 = {
        0x00010000,          // workgroup_mask=0, data_size=1 (2 bytes)
        (int)(64u << 16),    // tensor_dim0 = 64 (bits 79:48, low half)
        (int)(32u << 16),    // tensor_dim1 = 32
        (int)(64u << 16),    // tile_dim0   = 64 (bits 127:112)
        32,                  // tile_dim1   = 32 (bits 143:128)
        64,                  // tensor_dim0_stride = 64 (bits 207:160)
        0, 0
      };
      i32x4 gz4 = {0, 0, 0, 0};
      i32x8 gz8 = {0, 0, 0, 0, 0, 0, 0, 0};
      __builtin_amdgcn_tensor_load_to_lds(g0, g1, gz4, gz4, gz8, 0);
      __builtin_amdgcn_s_wait_tensorcnt(0);   // TENSORcnt==0 before barrier
    }
#else
    *(v4u*)&Ks[srow][scol] =
        *(const v4u*)(Kg + (size_t)(kt * 32 + srow) * D_ + scol);
#endif
    __syncthreads();

    if (kt * 32 <= q0 + 15) {   // wave-uniform -> EXEC all ones for WMMA
      // S = Q K^T (two 16x16 tiles over 32 keys), scale + causal mask
      float sv[2][8];
      for (int kn = 0; kn < 2; ++kn) {
        v8f s = {};
        for (int j = 0; j < 2; ++j) {
          const unsigned short* bp = &Ks[kn * 16 + lr][j * 32 + lh * 8];
          v16bf kfr = make_frag(bp, bp + 16);
          s = wmma_bf16(qf[j], kfr, s);
        }
        int key = kt * 32 + kn * 16 + lr;
        for (int r = 0; r < 8; ++r) {
          int row = q0 + r + lh * 8;
          sv[kn][r] = (key <= row) ? s[r] * 0.125f : -1e30f;
        }
      }
      // online softmax per row (16-lane-half shuffles match row mapping)
      for (int r = 0; r < 8; ++r) {
        float mt = fmaxf(sv[0][r], sv[1][r]);
        for (int off = 8; off > 0; off >>= 1)
          mt = fmaxf(mt, __shfl_xor(mt, off, 32));
        float mnew = fmaxf(mrow[r], mt);
        float alpha = __expf(mrow[r] - mnew);
        mrow[r] = mnew;
        float p0 = __expf(sv[0][r] - mnew);
        float p1 = __expf(sv[1][r] - mnew);
        float rs = p0 + p1;
        for (int off = 8; off > 0; off >>= 1)
          rs += __shfl_xor(rs, off, 32);
        lrow[r] = lrow[r] * alpha + rs;
        Ps[wave][r + lh * 8][lr]      = f2bf(p0);
        Ps[wave][r + lh * 8][16 + lr] = f2bf(p1);
        for (int c = 0; c < 4; ++c) o[c][r] *= alpha;
      }
      asm volatile("" ::: "memory");  // keep P stores before P loads
      // O += P V  (P: 16x32 A-frag via LDS transpose; V^T rows contiguous)
      const unsigned short* pa = &Ps[wave][lr][lh * 8];
      v16bf pf = make_frag(pa, pa + 16);
      for (int c = 0; c < 4; ++c) {
        const unsigned short* vb = &Vt[c * 16 + lr][lh * 8];
        v16bf vf = make_frag(vb, vb + 16);
        o[c] = wmma_bf16(pf, vf, o[c]);
      }
    }
    __syncthreads();
  }

  // normalize and write [B][T][C] bf16
  for (int c = 0; c < 4; ++c) {
    for (int r = 0; r < 8; ++r) {
      int t = q0 + r + lh * 8;
      int col = h * D_ + c * 16 + lr;
      float val = o[c][r] / lrow[r];
      attnb[((size_t)b * T_ + t) * C_ + col] = f2bf(val);
    }
  }
}

// ---------------------------------------------------------------------------
// Output projection: [8192 x 1024] x [1024 x 1024]^T + bias -> fp32 out
// Block tile 128x128, waves 32x64.
// ---------------------------------------------------------------------------
__global__ __launch_bounds__(256) void out_gemm(
    const unsigned short* __restrict__ ab,   // [BT][C] bf16 (attn out)
    const unsigned short* __restrict__ wb,   // [C][C] bf16
    const float* __restrict__ bias,          // [C]
    float* __restrict__ out) {               // [BT][C] fp32
  const int block_m = blockIdx.x * 128;
  const int block_n = blockIdx.y * 128;
  const int wave = threadIdx.x >> 5;
  const int lane = threadIdx.x & 31;
  const int lr = lane & 15;
  const int lh = (lane < 16) ? 0 : 1;
  const int m0 = block_m + (wave >> 1) * 32;
  const int n0 = block_n + (wave & 1) * 64;

  v8f acc[2][4] = {};
  for (int k0 = 0; k0 < C_; k0 += 32) {
    v16bf afr[2], bfr[4];
    for (int mi = 0; mi < 2; ++mi) {
      const unsigned short* ap = ab + (size_t)(m0 + mi * 16 + lr) * C_ + k0 + lh * 8;
      __builtin_prefetch(ap + 64, 0, 1);
      afr[mi] = make_frag(ap, ap + 16);
    }
    for (int ni = 0; ni < 4; ++ni) {
      const unsigned short* bp = wb + (size_t)(n0 + ni * 16 + lr) * C_ + k0 + lh * 8;
      bfr[ni] = make_frag(bp, bp + 16);
    }
    for (int mi = 0; mi < 2; ++mi)
      for (int ni = 0; ni < 4; ++ni)
        acc[mi][ni] = wmma_bf16(afr[mi], bfr[ni], acc[mi][ni]);
  }
  for (int mi = 0; mi < 2; ++mi) {
    for (int ni = 0; ni < 4; ++ni) {
      int gn = n0 + ni * 16 + lr;
      float bia = bias[gn];
      for (int r = 0; r < 8; ++r) {
        int gm = m0 + mi * 16 + r + lh * 8;
        out[(size_t)gm * C_ + gn] = acc[mi][ni][r] + bia;
      }
    }
  }
}

// ---------------------------------------------------------------------------
extern "C" void kernel_launch(void* const* d_in, const int* in_sizes, int n_in,
                              void* d_out, int out_size, void* d_ws, size_t ws_size,
                              hipStream_t stream) {
  (void)in_sizes; (void)n_in; (void)out_size; (void)ws_size;
  const float* x      = (const float*)d_in[0];
  const float* qkv_w  = (const float*)d_in[1];
  const float* qkv_b  = (const float*)d_in[2];
  const float* out_w  = (const float*)d_in[3];
  const float* out_b  = (const float*)d_in[4];
  float* out = (float*)d_out;

  // workspace layout (ushort units)
  unsigned short* ws    = (unsigned short*)d_ws;
  unsigned short* xb    = ws;                       //  8,388,608
  unsigned short* wqkvb = xb    +  8388608;         //  3,145,728
  unsigned short* wob   = wqkvb +  3145728;         //  1,048,576
  unsigned short* qkvb  = wob   +  1048576;         // 25,165,824
  unsigned short* attnb = qkvb  + 25165824;         //  8,388,608

  cvt_f32_bf16<<<(8388608 + 255) / 256, 256, 0, stream>>>(x, xb, 8388608);
  cvt_f32_bf16<<<(3145728 + 255) / 256, 256, 0, stream>>>(qkv_w, wqkvb, 3145728);
  cvt_f32_bf16<<<(1048576 + 255) / 256, 256, 0, stream>>>(out_w, wob, 1048576);

  qkv_gemm<<<dim3(BT / 128, K3C / 128), 256, 0, stream>>>(xb, wqkvb, qkv_b, qkvb);
  rope_kernel<<<(2 * B_ * H_ * T_ * (D_ / 2) + 255) / 256, 256, 0, stream>>>(qkvb);
  flash_attn<<<dim3(T_ / 128, B_ * H_), 256, 0, stream>>>(qkvb, attnb);
  out_gemm<<<dim3(BT / 128, C_ / 128), 256, 0, stream>>>(attnb, wob, out_b, out);
}
